// GatNet_56332791054870
// MI455X (gfx1250) — matrix operand order repaired and checked by reading, not compiled, not run on hardware
//
#include <hip/hip_runtime.h>
#include <cstring>

#define N_NODES 100000
#define N_EDGES 1600000
#define IN_CH   128
#define HIDDEN  64
#define HEADS   4

typedef __attribute__((ext_vector_type(16))) __bf16         v16bf;
typedef __attribute__((ext_vector_type(16))) unsigned short v16us;
typedef __attribute__((ext_vector_type(8)))  unsigned short v8us;
typedef __attribute__((ext_vector_type(8)))  float          v8f;

// fp32 -> bf16, round-to-nearest-even
__device__ __forceinline__ unsigned short f2bf(float f) {
    unsigned u = __builtin_bit_cast(unsigned, f);
    u += 0x7FFFu + ((u >> 16) & 1u);
    return (unsigned short)(u >> 16);
}
// order-preserving float<->uint for atomicMax on u32
__device__ __forceinline__ unsigned encf(float f) {
    unsigned u = __builtin_bit_cast(unsigned, f);
    return (u & 0x80000000u) ? ~u : (u | 0x80000000u);
}
__device__ __forceinline__ float decf(unsigned k) {
    unsigned u = (k & 0x80000000u) ? (k & 0x7FFFFFFFu) : ~k;
    return __builtin_bit_cast(float, u);
}
__device__ __forceinline__ float lrelu(float a) { return a > 0.0f ? a : 0.2f * a; }

// one-time fp32 -> packed bf16 conversion (x, x1, W1, W2)
__global__ void pack_bf16_k(const float* __restrict__ in,
                            unsigned short* __restrict__ out, int n) {
    int t = blockIdx.x * blockDim.x + threadIdx.x;
    if (t < n) out[t] = f2bf(in[t]);
}

// ---------------------------------------------------------------------------
// H[N x 64] = Xb[N x KD] @ Wb[64 x KD]^T  (pre-packed bf16, f32 accumulate)
// grid.x = N/16 M-tiles, block = 128 (wave w owns N-tile w). KD compile-time
// -> fully unrolled K loop: clause of b128 loads + back-to-back v_wmma chain.
// All lane fragment pieces are contiguous 16B-aligned 8xbf16 chunks:
//   A (16x32): lane l = row, half0 -> K {0..7,16..23}, half1 -> {8..15,24..31}
//   B (32x16): lane l = column, half0 -> K {0..7,16..23}, half1 -> {8..15,24..31}
// No divergence before WMMA (EXEC all-ones, ISA 7.12); 100000 = 6250*16.
// ---------------------------------------------------------------------------
template <int KD>
__global__ void __launch_bounds__(128)
gemm_bf16_wmma(const unsigned short* __restrict__ Xb,
               const unsigned short* __restrict__ Wb,
               float* __restrict__ H) {
    const int wave = threadIdx.x >> 5;   // N-tile 0..3 (HIDDEN = 64)
    const int lane = threadIdx.x & 31;
    const int half = lane >> 4;
    const int l    = lane & 15;

    const unsigned short* arow = Xb + (long)(blockIdx.x * 16 + l) * KD + half * 8;
    const unsigned short* brow = Wb + (long)(wave * 16 + l) * KD + half * 8;

    v8f acc = {};
#pragma unroll
    for (int k0 = 0; k0 < KD; k0 += 32) {
        union { v16us u; v8us h2[2]; v16bf b; } A, B;
        A.h2[0] = *(const v8us*)(arow + k0);        // K = k0 + half*8 + 0..7
        A.h2[1] = *(const v8us*)(arow + k0 + 16);   // K = k0 + 16 + half*8 + 0..7
        B.h2[0] = *(const v8us*)(brow + k0);        // K = k0 + half*8 + 0..7
        B.h2[1] = *(const v8us*)(brow + k0 + 16);   // K = k0 + 16 + half*8 + 0..7
        acc = __builtin_amdgcn_wmma_f32_16x16x32_bf16(
                  false, A.b, false, B.b, (short)0, acc, false, false);
    }
    // D: VGPR r -> row half*8 + r, column = lane&15 within tile
    float* outp = H + (long)(blockIdx.x * 16 + half * 8) * HIDDEN + wave * 16 + l;
#pragma unroll
    for (int r = 0; r < 8; ++r)
        outp[(long)r * HIDDEN] = acc[r];
}

// e_src[n,h] = <h[n,head], a_src[head]>, e_dst likewise. thread = node*4+head
__global__ void attn_coef(const float* __restrict__ h,
                          const float* __restrict__ a_src,
                          const float* __restrict__ a_dst,
                          float* __restrict__ e_src, float* __restrict__ e_dst) {
    int t = blockIdx.x * blockDim.x + threadIdx.x;
    if (t >= N_NODES * HEADS) return;
    int head = t & 3, node = t >> 2;
    const float* hp = h + (long)node * HIDDEN + head * 16;
    float s = 0.f, d = 0.f;
#pragma unroll
    for (int c = 0; c < 16; ++c) {
        float v = hp[c];
        s += v * a_src[head * 16 + c];
        d += v * a_dst[head * 16 + c];
    }
    e_src[t] = s; e_dst[t] = d;
}

__global__ void fill_f32(float* p, float v, int n) {
    int t = blockIdx.x * blockDim.x + threadIdx.x;
    if (t < n) p[t] = v;
}
__global__ void fill_u32(unsigned* p, unsigned v, int n) {
    int t = blockIdx.x * blockDim.x + threadIdx.x;
    if (t < n) p[t] = v;
}

__device__ __forceinline__ void edge_ids(int e, const int* src, const int* dst,
                                         int& s, int& d) {
    if (e < N_EDGES) { s = src[e]; d = dst[e]; }
    else             { s = d = e - N_EDGES; }          // appended self-loops
}

// segment max: thread = edge*4 + head, native GLOBAL_ATOMIC_MAX_U32
__global__ void edge_max_k(const int* __restrict__ src, const int* __restrict__ dst,
                           const float* __restrict__ e_src, const float* __restrict__ e_dst,
                           unsigned* __restrict__ amax) {
    const int EP = N_EDGES + N_NODES;
    int t = blockIdx.x * blockDim.x + threadIdx.x;
    if (t >= EP * HEADS) return;
    int head = t & 3, e = t >> 2, s, d;
    edge_ids(e, src, dst, s, d);
    float a = lrelu(e_src[s * 4 + head] + e_dst[d * 4 + head]);
    atomicMax(&amax[d * 4 + head], encf(a));
}

// segment sum of exp(alpha - max): native GLOBAL_ATOMIC_ADD_F32
__global__ void edge_sum_k(const int* __restrict__ src, const int* __restrict__ dst,
                           const float* __restrict__ e_src, const float* __restrict__ e_dst,
                           const unsigned* __restrict__ amax, float* __restrict__ denom) {
    const int EP = N_EDGES + N_NODES;
    int t = blockIdx.x * blockDim.x + threadIdx.x;
    if (t >= EP * HEADS) return;
    int head = t & 3, e = t >> 2, s, d;
    edge_ids(e, src, dst, s, d);
    float a = lrelu(e_src[s * 4 + head] + e_dst[d * 4 + head]);
    atomicAdd(&denom[d * 4 + head], __expf(a - decf(amax[d * 4 + head])));
}

// aggregation: thread = edge*64 + channel -> coalesced gather of h[src] and
// coalesced atomic scatter into agg[dst]; alpha recomputed (L2-hot arrays)
__global__ void edge_agg_k(const int* __restrict__ src, const int* __restrict__ dst,
                           const float* __restrict__ e_src, const float* __restrict__ e_dst,
                           const unsigned* __restrict__ amax, const float* __restrict__ denom,
                           const float* __restrict__ h, float* __restrict__ agg) {
    const int EP = N_EDGES + N_NODES;
    int t = blockIdx.x * blockDim.x + threadIdx.x;
    if (t >= EP * HIDDEN) return;
    int c = t & 63, e = t >> 6, head = c >> 4, s, d;
    edge_ids(e, src, dst, s, d);
    float a = lrelu(e_src[s * 4 + head] + e_dst[d * 4 + head]);
    float w = __expf(a - decf(amax[d * 4 + head]));
    float coeff = w / (denom[d * 4 + head] + 1e-16f);
    atomicAdd(&agg[(long)d * HIDDEN + c], h[(long)s * HIDDEN + c] * coeff);
}

// x = elu(agg + bias), in place
__global__ void bias_elu(float* __restrict__ agg, const float* __restrict__ b) {
    int t = blockIdx.x * blockDim.x + threadIdx.x;
    if (t >= N_NODES * HIDDEN) return;
    float v = agg[t] + b[t & 63];
    agg[t] = v > 0.0f ? v : (__expf(v) - 1.0f);
}

// out[n] = <x2[n], fc_w> + fc_b
__global__ void fc_kernel(const float* __restrict__ X, const float* __restrict__ W,
                          const float* __restrict__ B, float* __restrict__ out) {
    int i = blockIdx.x * blockDim.x + threadIdx.x;
    if (i >= N_NODES) return;
    const float4* xp = (const float4*)(X + (long)i * HIDDEN);
    const float4* wp = (const float4*)W;
    float acc = 0.f;
#pragma unroll
    for (int j = 0; j < HIDDEN / 4; ++j) {
        float4 a = xp[j], b = wp[j];
        acc += a.x * b.x + a.y * b.y + a.z * b.z + a.w * b.w;
    }
    out[i] = acc + B[0];
}

extern "C" void kernel_launch(void* const* d_in, const int* in_sizes, int n_in,
                              void* d_out, int out_size, void* d_ws, size_t ws_size,
                              hipStream_t stream) {
    const float* x      = (const float*)d_in[0];
    const int*   ei     = (const int*)  d_in[1];
    const float* w1     = (const float*)d_in[2];
    const float* a_src1 = (const float*)d_in[3];
    const float* a_dst1 = (const float*)d_in[4];
    const float* b1     = (const float*)d_in[5];
    const float* w2     = (const float*)d_in[6];
    const float* a_src2 = (const float*)d_in[7];
    const float* a_dst2 = (const float*)d_in[8];
    const float* b2     = (const float*)d_in[9];
    const float* fc_w   = (const float*)d_in[10];
    const float* fc_b   = (const float*)d_in[11];
    const int* src = ei;
    const int* dst = ei + N_EDGES;

    // workspace layout:
    //  f32:  h[N*64] | agg[N*64] | e_src[N*4] | e_dst[N*4] | denom[N*4] | amax[N*4 u32]
    //  bf16: xb[N*128] | wb[64*128]
    float* ws    = (float*)d_ws;
    float* h     = ws;
    float* agg   = h     + (size_t)N_NODES * HIDDEN;   // also serves as x1/x2
    float* e_src = agg   + (size_t)N_NODES * HIDDEN;
    float* e_dst = e_src + (size_t)N_NODES * HEADS;
    float* denom = e_dst + (size_t)N_NODES * HEADS;
    unsigned* amax = (unsigned*)(denom + (size_t)N_NODES * HEADS);
    unsigned short* xb = (unsigned short*)(amax + (size_t)N_NODES * HEADS);
    unsigned short* wb = xb + (size_t)N_NODES * IN_CH;

    float negmax = -3.402823466e38f;
    unsigned encneg; std::memcpy(&encneg, &negmax, 4);
    encneg = (encneg & 0x80000000u) ? ~encneg : (encneg | 0x80000000u);

    const int EP  = N_EDGES + N_NODES;
    const int NH  = N_NODES * HEADS;                   // 400k
    const int NC  = N_NODES * HIDDEN;                  // 6.4M
    const int EH  = EP * HEADS;                        // 6.8M
    const int EC  = EP * HIDDEN;                       // 108.8M
    auto blk = [](int n, int b) { return (n + b - 1) / b; };

    for (int layer = 0; layer < 2; ++layer) {
        const float* W  = layer ? w2 : w1;
        const float* as = layer ? a_src2 : a_src1;
        const float* ad = layer ? a_dst2 : a_dst1;
        const float* bb = layer ? b2 : b1;
        const float* Xin = layer ? agg : x;            // layer2 input x1 lives in agg
        const int Kdim   = layer ? HIDDEN : IN_CH;

        // one-time bf16 packing of activations + weights, then WMMA GEMM
        pack_bf16_k<<<blk(N_NODES * Kdim, 256), 256, 0, stream>>>(Xin, xb, N_NODES * Kdim);
        pack_bf16_k<<<blk(HIDDEN * Kdim, 256), 256, 0, stream>>>(W, wb, HIDDEN * Kdim);
        if (layer == 0)
            gemm_bf16_wmma<IN_CH><<<N_NODES / 16, 128, 0, stream>>>(xb, wb, h);
        else
            gemm_bf16_wmma<HIDDEN><<<N_NODES / 16, 128, 0, stream>>>(xb, wb, h);

        // per-node attention coefficients
        attn_coef<<<blk(NH, 256), 256, 0, stream>>>(h, as, ad, e_src, e_dst);
        // init reductions (agg init overwrites x1 in layer 2 — x1 is dead here)
        fill_u32<<<blk(NH, 256), 256, 0, stream>>>(amax, encneg, NH);
        fill_f32<<<blk(NH, 256), 256, 0, stream>>>(denom, 0.0f, NH);
        fill_f32<<<blk(NC, 256), 256, 0, stream>>>(agg, 0.0f, NC);
        // segment softmax + aggregation over 1.7M edges
        edge_max_k<<<blk(EH, 256), 256, 0, stream>>>(src, dst, e_src, e_dst, amax);
        edge_sum_k<<<blk(EH, 256), 256, 0, stream>>>(src, dst, e_src, e_dst, amax, denom);
        edge_agg_k<<<blk(EC, 256), 256, 0, stream>>>(src, dst, e_src, e_dst, amax,
                                                     denom, h, agg);
        // x_{l+1} = elu(agg + bias), in place
        bias_elu<<<blk(NC, 256), 256, 0, stream>>>(agg, bb);
    }
    // out = x2 @ fc_w^T + fc_b
    fc_kernel<<<blk(N_NODES, 256), 256, 0, stream>>>(agg, fc_w, fc_b, (float*)d_out);
}
